// SingleAttentionHead_66881230733413
// MI455X (gfx1250) — compile-verified, matrix operand order checked by reading, not run on hardware
//
#include <hip/hip_runtime.h>
#include <hip/hip_bf16.h>
#include <math.h>

typedef __attribute__((ext_vector_type(16))) _Float16 v16h;
typedef __attribute__((ext_vector_type(8)))  _Float16 v8h;
typedef __attribute__((ext_vector_type(8)))  float    v8f;

#define DEV __device__ __forceinline__

constexpr int BATCH = 8;
constexpr int SEQ   = 2048;
constexpr int DIM   = 512;
constexpr int MROWS = BATCH * SEQ;                   // 16384 projection rows
constexpr float SM_SCALE = 0.04419417382415922f;     // 1/sqrt(512)

// LDS layout for the flash kernel (dynamic shared memory)
constexpr int QROW_STRIDE = 520;   // halves; 1040B -> banks 4i%64, conflict-free
constexpr int PROW_STRIDE = 72;    // halves; 144B  -> banks 36i%64, conflict-free
constexpr int LDS_Q_BYTES = 64 * QROW_STRIDE * 2;            // 66560
constexpr int LDS_P_BYTES = 4 * 16 * PROW_STRIDE * 2;        // 9216
constexpr int LDS_S_BYTES = 2 * 2 * 4 * 4 * 16 * 4;          // 4096
constexpr int FLASH_LDS   = LDS_Q_BYTES + LDS_P_BYTES + LDS_S_BYTES;

// v16h element i maps to K = (i<8 ? i : 8+i) + hid*8  -> two contiguous
// 8-element (16B) segments at +0 and +16 (ISA 7.12.2 16-bit A/B layout).
DEV v16h load16(const _Float16* p) {   // p already includes hid*8 offset
  const v8h lo = *(const v8h*)p;
  const v8h hi = *(const v8h*)(p + 16);
  v16h r;
#pragma unroll
  for (int i = 0; i < 8; ++i) { r[i] = lo[i]; r[i + 8] = hi[i]; }
  return r;
}

DEV v8f zero8() {
  v8f z;
#pragma unroll
  for (int i = 0; i < 8; ++i) z[i] = 0.0f;
  return z;
}

// reductions across a 16-lane half-wave (rows of the C layout)
DEV float halfmax(float v) {
#pragma unroll
  for (int m = 1; m < 16; m <<= 1) v = fmaxf(v, __shfl_xor(v, m, 32));
  return v;
}
DEV float halfsum(float v) {
#pragma unroll
  for (int m = 1; m < 16; m <<= 1) v += __shfl_xor(v, m, 32);
  return v;
}

// ---------------------------------------------------------------------------
// Weight convert+transpose: Wt[n][k] (f16) = W[k][n] (f32). 512x512.
// ---------------------------------------------------------------------------
__global__ __launch_bounds__(256)
void wcvt_kernel(const float* __restrict__ W, _Float16* __restrict__ Wt) {
  int idx = blockIdx.x * 256 + threadIdx.x;      // grid = 512*512/256
  int k = idx >> 9;
  int n = idx & 511;
  Wt[(size_t)n * DIM + k] = (_Float16)W[(size_t)k * DIM + n];
}

// ---------------------------------------------------------------------------
// Projection: Y = X[f32, MROWSxDIM] @ W  (Wt is pre-transposed f16).
// block = 128 (4 waves); wave: 16 rows x 64 cols. grid = (MROWS/16, DIM/256).
// TRANSPOSE_OUT (compile-time): store Vt[b][n][k] instead of row-major.
// ---------------------------------------------------------------------------
template <int TRANSPOSE_OUT>
__global__ __launch_bounds__(128)
void proj_f16_kernel(const float* __restrict__ X, const _Float16* __restrict__ Wt,
                     _Float16* __restrict__ Y) {
  const int lane  = threadIdx.x & 31;
  const int wave  = threadIdx.x >> 5;
  const int l16   = lane & 15;
  const int hid   = lane >> 4;
  const int rbase = blockIdx.x * 16;
  const int nbase = blockIdx.y * 256 + wave * 64;

  const float* xrow = X + (size_t)(rbase + l16) * DIM + hid * 8;
  const _Float16* wrow[4];
#pragma unroll
  for (int nt = 0; nt < 4; ++nt)
    wrow[nt] = Wt + (size_t)(nbase + nt * 16 + l16) * DIM + hid * 8;

  auto loadA = [&](int kb) {
    const float* xp = xrow + kb * 32;
    float4 a0 = *(const float4*)(xp);
    float4 a1 = *(const float4*)(xp + 4);
    float4 b0 = *(const float4*)(xp + 16);
    float4 b1 = *(const float4*)(xp + 20);
    v16h a;
    a[0] = (_Float16)a0.x;  a[1] = (_Float16)a0.y;
    a[2] = (_Float16)a0.z;  a[3] = (_Float16)a0.w;
    a[4] = (_Float16)a1.x;  a[5] = (_Float16)a1.y;
    a[6] = (_Float16)a1.z;  a[7] = (_Float16)a1.w;
    a[8] = (_Float16)b0.x;  a[9] = (_Float16)b0.y;
    a[10] = (_Float16)b0.z; a[11] = (_Float16)b0.w;
    a[12] = (_Float16)b1.x; a[13] = (_Float16)b1.y;
    a[14] = (_Float16)b1.z; a[15] = (_Float16)b1.w;
    return a;
  };

  v8f acc[4];
#pragma unroll
  for (int t = 0; t < 4; ++t) acc[t] = zero8();

  v16h a = loadA(0);
#pragma unroll
  for (int kb = 0; kb < DIM / 32; ++kb) {
    v16h bcur = load16(wrow[0] + kb * 32);
    v16h aN;
    if (kb < DIM / 32 - 1) aN = loadA(kb + 1);
#pragma unroll
    for (int nt = 0; nt < 4; ++nt) {
      v16h bN;
      if (nt < 3) bN = load16(wrow[nt + 1] + kb * 32);
      acc[nt] = __builtin_amdgcn_wmma_f32_16x16x32_f16(
          false, a, false, bcur, (short)0, acc[nt], false, false);
      if (nt < 3) bcur = bN;
    }
    if (kb < DIM / 32 - 1) a = aN;
  }

  // C layout: vgpr r -> rows r / r+8 across lane halves
#pragma unroll
  for (int nt = 0; nt < 4; ++nt) {
#pragma unroll
    for (int r = 0; r < 8; ++r) {
      int row = rbase + r + hid * 8;
      int col = nbase + nt * 16 + l16;
      if (TRANSPOSE_OUT) {
        // Vt[b][col][k], b = row/SEQ, k = row%SEQ
        Y[((size_t)((row >> 11) * DIM + col)) * SEQ + (row & (SEQ - 1))] =
            (_Float16)acc[nt][r];
      } else {
        Y[(size_t)row * DIM + col] = (_Float16)acc[nt][r];
      }
    }
  }
}

// ---------------------------------------------------------------------------
// Fused causal flash attention.
// block = 512 threads (16 waves) per 64-query tile, grid = (SEQ/64, BATCH).
// wave = qt*4 + cg. Q tile staged once in LDS (padded rows, conflict-free);
// score phase: wave (qt,cg) computes full-D 16x16 tile (A from LDS, B=K rows
// from cache, 4-way shared across qt-waves). P@V: A=P tile from LDS, B=Vt
// rows from cache (4-way shared). Per-qt stats double-buffered in LDS.
// ---------------------------------------------------------------------------
__global__ __launch_bounds__(512)
void flash_attn_kernel(const _Float16* __restrict__ Qf,
                       const _Float16* __restrict__ Kf,
                       const _Float16* __restrict__ Vt,
                       float* __restrict__ out) {
  extern __shared__ __align__(16) unsigned char smem[];
  _Float16* lds_q = (_Float16*)smem;                          // 65 KB padded
  _Float16* lds_p = (_Float16*)(smem + LDS_Q_BYTES);          // 9 KB padded
  float* lds_stats = (float*)(smem + LDS_Q_BYTES + LDS_P_BYTES);

  const int tid  = threadIdx.x;
  const int lane = tid & 31;
  const int wave = tid >> 5;
  const int l16  = lane & 15;
  const int hid  = lane >> 4;
  const int qt   = wave >> 2;
  const int cg   = wave & 3;
  const int qb   = blockIdx.x * 64;
  const int b    = blockIdx.y;
  const int qtb  = qb + qt * 16;       // this wave's query rows base

  // ---- stage the block's Q tile (64 x 512 f16, contiguous) into LDS ----
  {
    const uint4* src = (const uint4*)(Qf + (size_t)(b * SEQ + qb) * DIM);
    uint4* dst = (uint4*)lds_q;
#pragma unroll
    for (int i = tid; i < 64 * DIM / 8; i += 512) {
      int row = i >> 6, seg = i & 63;                 // 64 x 16B per row
      dst[row * (QROW_STRIDE / 8) + seg] = src[i];
    }
  }
  __syncthreads();

  const _Float16* qrow  = lds_q + (qt * 16 + l16) * QROW_STRIDE + hid * 8;
  const _Float16* krow0 = Kf + (size_t)(b * SEQ + cg * 16 + l16) * DIM + hid * 8;
  const _Float16* vbase = Vt + (size_t)(b * DIM + cg * 128 + l16) * SEQ;
  const _Float16* prow  = lds_p + qt * 16 * PROW_STRIDE + l16 * PROW_STRIDE + hid * 8;

  v8f o[8];
#pragma unroll
  for (int nt = 0; nt < 8; ++nt) o[nt] = zero8();
  float mrow[8], lrow[8];
#pragma unroll
  for (int r = 0; r < 8; ++r) { mrow[r] = -INFINITY; lrow[r] = 0.0f; }

  const int kend = qb + 64;            // causal: keys 0 .. qb+63 (<= SEQ)
  int parity = 0;
  for (int kb = 0; kb < kend; kb += 64, parity ^= 1) {
    const _Float16* krow = krow0 + (size_t)kb * DIM;

    // prefetch next chunk's K rows for this wave's key subtile
    if (kb + 64 < kend) __builtin_prefetch(krow + (size_t)64 * DIM);

    // ---- scores: full-D 16x16 tile (A from LDS, B from cache) ----
    v8f sc = zero8();
    v16h qa = load16(qrow);
    v16h km = load16(krow);
#pragma unroll
    for (int c = 0; c < 16; ++c) {
      v16h qaN, kmN;
      if (c < 15) {
        qaN = load16(qrow + (c + 1) * 32);
        kmN = load16(krow + (c + 1) * 32);
      }
      sc = __builtin_amdgcn_wmma_f32_16x16x32_f16(
          false, qa, false, km, (short)0, sc, false, false);
      if (c < 15) { qa = qaN; km = kmN; }
    }

    // ---- scale + causal mask, local stats, unnormalized p ----
    const int kidx = kb + cg * 16 + l16;
    float s[8], p[8], mw[8], lw[8];
#pragma unroll
    for (int r = 0; r < 8; ++r) {
      int qidx = qtb + r + 8 * hid;
      s[r] = (kidx > qidx) ? -INFINITY : sc[r] * SM_SCALE;
      mw[r] = halfmax(s[r]);
      bool dead = (mw[r] == -INFINITY);          // fully masked tile row
      p[r] = dead ? 0.0f : __expf(s[r] - mw[r]);
      lw[r] = halfsum(p[r]);
    }
    // ---- publish stats (one lane per half-wave) ----
    if (l16 == 0) {
#pragma unroll
      for (int r = 0; r < 8; ++r) {
        int row = 8 * hid + r;
        lds_stats[(((parity * 2 + 0) * 4 + qt) * 4 + cg) * 16 + row] = mw[r];
        lds_stats[(((parity * 2 + 1) * 4 + qt) * 4 + cg) * 16 + row] = lw[r];
      }
    }
    __syncthreads();

    // ---- merge stats across the 4 cg-waves of this qt ----
    float corr[8], fself[8];
#pragma unroll
    for (int r = 0; r < 8; ++r) {
      int row = 8 * hid + r;
      float mmax = mrow[r];
#pragma unroll
      for (int g = 0; g < 4; ++g)
        mmax = fmaxf(mmax,
            lds_stats[(((parity * 2 + 0) * 4 + qt) * 4 + g) * 16 + row]);
      float lsum = 0.0f;
#pragma unroll
      for (int g = 0; g < 4; ++g) {
        float mwv = lds_stats[(((parity * 2 + 0) * 4 + qt) * 4 + g) * 16 + row];
        float lwv = lds_stats[(((parity * 2 + 1) * 4 + qt) * 4 + g) * 16 + row];
        lsum += (mwv == -INFINITY ? 0.0f : __expf(mwv - mmax)) * lwv;
      }
      corr[r]  = __expf(mrow[r] - mmax);         // first iter: exp(-inf)=0
      fself[r] = (mw[r] == -INFINITY) ? 0.0f : __expf(mw[r] - mmax);
      lrow[r]  = lrow[r] * corr[r] + lsum;
      mrow[r]  = mmax;
    }
    // ---- write this wave's scaled P (f16) into [qt][16][64pad] tile ----
#pragma unroll
    for (int r = 0; r < 8; ++r)
      lds_p[(qt * 16 + 8 * hid + r) * PROW_STRIDE + cg * 16 + l16] =
          (_Float16)(p[r] * fself[r]);
    __syncthreads();

    // ---- rescale O, then O += P @ V (pipelined over nt) ----
#pragma unroll
    for (int nt = 0; nt < 8; ++nt)
#pragma unroll
      for (int r = 0; r < 8; ++r)
        o[nt][r] *= corr[r];

#pragma unroll
    for (int kc = 0; kc < 2; ++kc) {
      v16h pa = load16(prow + kc * 32);
      const int s0 = kb + kc * 32 + hid * 8;     // always in-bounds
      v16h vb = load16(vbase + s0);
#pragma unroll
      for (int nt = 0; nt < 8; ++nt) {
        v16h vbN;
        if (nt < 7) vbN = load16(vbase + (nt + 1) * 16 * SEQ + s0);
        o[nt] = __builtin_amdgcn_wmma_f32_16x16x32_f16(
            false, pa, false, vb, (short)0, o[nt], false, false);
        if (nt < 7) vb = vbN;
      }
    }
    // no trailing barrier: next iteration's stats barrier (parity-flipped
    // stats buffer) orders lds_p reuse.
  }

  // ---- normalize and write f32 output ----
  float* orow = out + (size_t)(b * SEQ + qtb) * DIM;
#pragma unroll
  for (int r = 0; r < 8; ++r) {
    float inv = 1.0f / lrow[r];
    float* op = orow + (r + 8 * hid) * DIM + cg * 128 + l16;
#pragma unroll
    for (int nt = 0; nt < 8; ++nt) op[nt * 16] = o[nt][r] * inv;
  }
}

// ---------------------------------------------------------------------------
extern "C" void kernel_launch(void* const* d_in, const int* in_sizes, int n_in,
                              void* d_out, int out_size, void* d_ws, size_t ws_size,
                              hipStream_t stream) {
  const float* Xk = (const float*)d_in[0];   // inputs_for_keys
  const float* Xv = (const float*)d_in[1];   // inputs_for_values
  const float* Xq = (const float*)d_in[2];   // inputs_for_queries
  const float* Wq = (const float*)d_in[3];
  const float* Wk = (const float*)d_in[4];
  const float* Wv = (const float*)d_in[5];

  _Float16* Qf = (_Float16*)d_ws;                       // 16 MB, [b*S+s][d]
  _Float16* Kf = Qf + (size_t)MROWS * DIM;              // 16 MB, [b*S+s][d]
  _Float16* Vt = Kf + (size_t)MROWS * DIM;              // 16 MB, [b][d][s]
  _Float16* Wt = Vt + (size_t)MROWS * DIM;              // 512 KB, reused x3

  dim3 pblk(128);
  dim3 pgrid(MROWS / 16, DIM / 256);
  dim3 wgrid(DIM * DIM / 256);

  wcvt_kernel<<<wgrid, 256, 0, stream>>>(Wq, Wt);
  proj_f16_kernel<0><<<pgrid, pblk, 0, stream>>>(Xq, Wt, Qf);
  wcvt_kernel<<<wgrid, 256, 0, stream>>>(Wk, Wt);
  proj_f16_kernel<0><<<pgrid, pblk, 0, stream>>>(Xk, Wt, Kf);
  wcvt_kernel<<<wgrid, 256, 0, stream>>>(Wv, Wt);
  proj_f16_kernel<1><<<pgrid, pblk, 0, stream>>>(Xv, Wt, Vt);

  dim3 fgrid(SEQ / 64, BATCH);
  flash_attn_kernel<<<fgrid, 512, FLASH_LDS, stream>>>(Qf, Kf, Vt,
                                                       (float*)d_out);
}
